// Self_Attention_Layer_16999480558329
// MI455X (gfx1250) — matrix operand order repaired and checked by reading, not compiled, not run on hardware
//
#include <hip/hip_runtime.h>
#include <hip/hip_bf16.h>

// ---------------------------------------------------------------------------
// Self-attention, MI455X (gfx1250), bf16 WMMA + TDM + async-LDS pipeline.
//   k0a: X fp32 -> bf16 (once)
//   k0b: W fp32 [k][n] -> bf16 transposed [n][k] (once per weight)
//   k1 (x3): projection GEMM, all-bf16, async global->LDS tiles
//   k2: flash attention; K/V tiles via Tensor Data Mover, L2 prefetch overlap
//   k3: output GEMM + bias, async tiles, fp32 out
// ---------------------------------------------------------------------------

#if __has_builtin(__builtin_amdgcn_tensor_load_to_lds)
#define HAVE_TDM 1
#else
#define HAVE_TDM 0
#endif

typedef __bf16 bf16_t;
typedef __attribute__((ext_vector_type(16))) __bf16 v16bf;
typedef __attribute__((ext_vector_type(8)))  __bf16 v8bf;
typedef __attribute__((ext_vector_type(8)))  float  v8f;
typedef __attribute__((ext_vector_type(4)))  float  v4f;

static __device__ __forceinline__ v16bf cat16(v8bf lo, v8bf hi) {
    return __builtin_shufflevector(lo, hi, 0,1,2,3,4,5,6,7,8,9,10,11,12,13,14,15);
}

static __device__ __forceinline__ unsigned int pack_bf2(float a, float b) {
    unsigned short ua = __builtin_bit_cast(unsigned short, (bf16_t)a);
    unsigned short ub = __builtin_bit_cast(unsigned short, (bf16_t)b);
    return (unsigned int)ua | ((unsigned int)ub << 16);
}

#define WMMA_BF16(A, B, C) \
    __builtin_amdgcn_wmma_f32_16x16x32_bf16(false, (A), false, (B), (short)0, (C), false, false)

// async 16B global -> LDS copy (ASYNCcnt-tracked)
static __device__ __forceinline__ void async_b128(void* lds_dst, const void* gsrc) {
    unsigned int dst = (unsigned int)(size_t)lds_dst;
    unsigned long long src = (unsigned long long)(size_t)gsrc;
    asm volatile("global_load_async_to_lds_b128 %0, %1, off"
                 :: "v"(dst), "v"(src) : "memory");
}
#define WAIT_ASYNC() asm volatile("s_wait_asynccnt 0" ::: "memory")

// ---------------------------------------------------------------------------
// Tensor Data Mover: 2D bf16 tile load, descriptor per CDNA5 ISA.
// ---------------------------------------------------------------------------
#if HAVE_TDM
typedef __attribute__((ext_vector_type(4))) unsigned int u32x4;
typedef __attribute__((ext_vector_type(8))) int          i32x8;
typedef __attribute__((ext_vector_type(4))) int          i32x4;

static __device__ __forceinline__ void tdm_load_2d_bf16(
    unsigned int lds_addr, const void* gptr, unsigned int d0flags,
    unsigned int tensor_d0, unsigned int tensor_d1,
    unsigned int tile_d0, unsigned int tile_d1,
    unsigned long long stride_elems)
{
    unsigned long long ga = (unsigned long long)(size_t)gptr;
    u32x4 g0 = { 1u, lds_addr, (unsigned int)ga,
                 (unsigned int)(ga >> 32) | (2u << 30) };
    i32x8 g1;
    g1[0] = (int)d0flags;
    g1[1] = (int)(tensor_d0 << 16);
    g1[2] = (int)((tensor_d0 >> 16) | (tensor_d1 << 16));
    g1[3] = (int)((tensor_d1 >> 16) | (tile_d0 << 16));
    g1[4] = (int)(tile_d1 & 0xFFFFu);
    g1[5] = (int)(unsigned int)stride_elems;
    g1[6] = (int)(unsigned int)(stride_elems >> 32);
    g1[7] = 0;
    i32x4 gz = {0, 0, 0, 0};
#if defined(__clang_major__) && __clang_major__ >= 23
    i32x8 gz8 = {0, 0, 0, 0, 0, 0, 0, 0};
    __builtin_amdgcn_tensor_load_to_lds(g0, g1, gz, gz, gz8, 0);
#else
    __builtin_amdgcn_tensor_load_to_lds(g0, g1, gz, gz, 0);
#endif
}
#define TDM_FLAGS_K ((1u<<16) | (1u<<20) | (6u<<22) | (3u<<25))  // pad 4DW/512B
#define TDM_FLAGS_V ((1u<<16) | (1u<<20) | (3u<<22) | (3u<<25))  // pad 4DW/64B
#endif

// ---------------------------------------------------------------------------
// Kernel 0a: fp32 -> bf16 bulk convert (chunks of 8)
// ---------------------------------------------------------------------------
__global__ __launch_bounds__(256) void cvt_bf16_kernel(
    const float* __restrict__ in, bf16_t* __restrict__ out)
{
    size_t idx = (size_t)blockIdx.x * 256 + threadIdx.x;
    const float* p = in + idx * 8;
    v4f a = *(const v4f*)p;
    v4f b = *(const v4f*)(p + 4);
    v8bf o;
    o[0] = (bf16_t)a.x; o[1] = (bf16_t)a.y; o[2] = (bf16_t)a.z; o[3] = (bf16_t)a.w;
    o[4] = (bf16_t)b.x; o[5] = (bf16_t)b.y; o[6] = (bf16_t)b.z; o[7] = (bf16_t)b.w;
    *(v8bf*)(out + idx * 8) = o;
}

// ---------------------------------------------------------------------------
// Kernel 0b: W fp32 [k][n] -> WT bf16 [n][k] (64x64 LDS-tiled transpose)
// ---------------------------------------------------------------------------
__global__ __launch_bounds__(256) void wt_transpose_kernel(
    const float* __restrict__ W, bf16_t* __restrict__ WT)
{
    __shared__ __align__(16) bf16_t T[64 * 72];
    const int tid = threadIdx.x;
    const int k0 = blockIdx.y * 64, n0 = blockIdx.x * 64;
#pragma unroll
    for (int i = 0; i < 16; ++i) {
        int idx = tid + i * 256;
        int kl = idx >> 6, nl = idx & 63;
        T[nl * 72 + kl] = (bf16_t)W[(size_t)(k0 + kl) * 1024 + n0 + nl];
    }
    __syncthreads();
#pragma unroll
    for (int i = 0; i < 2; ++i) {
        int c = tid + i * 256;
        int nl = c >> 3, ch = c & 7;
        *(v8bf*)(WT + (size_t)(n0 + nl) * 1024 + k0 + ch * 8) =
            *(const v8bf*)(T + nl * 72 + ch * 8);
    }
}

// ---------------------------------------------------------------------------
// Kernel 1: C[16384,1024] = A(bf16) @ Bt(bf16,[n][k])^T-view, output bf16.
// MODE 0: row-major (Q, K).  MODE 1: V transposed [B,H,dh,S].
// Tiles arrive via global_load_async_to_lds_b128.
// ---------------------------------------------------------------------------
template<int MODE>
__global__ __launch_bounds__(256) void proj_gemm_kernel(
    const bf16_t* __restrict__ A, const bf16_t* __restrict__ Bt,
    bf16_t* __restrict__ out)
{
    constexpr int Kdim = 1024, N = 1024;
    constexpr int LDA = 40, LDB = 40, LDC = 136;
    __shared__ __align__(16) bf16_t sm[128 * LDC];
    bf16_t* As = sm;
    bf16_t* Bs = sm + 128 * LDA;
    bf16_t* Cs = sm;

    const int tid  = threadIdx.x;
    const int lane = tid & 31, w = tid >> 5;
    const int g    = lane >> 4, r16 = lane & 15;
    const int wr   = w >> 2,  wc  = w & 3;
    const int bm   = blockIdx.y * 128, bn = blockIdx.x * 128;

    const v8f z8 = {0,0,0,0,0,0,0,0};
    v8f acc[4][2];
#pragma unroll
    for (int a = 0; a < 4; ++a)
#pragma unroll
        for (int b = 0; b < 2; ++b) acc[a][b] = z8;

    for (int kk = 0; kk < Kdim; kk += 32) {
        __syncthreads();
#pragma unroll
        for (int i = 0; i < 2; ++i) {          // A tile 128x32
            int idx = tid + i * 256;
            int m = idx >> 2, ch = idx & 3;
            async_b128(As + m * LDA + ch * 8,
                       A + (size_t)(bm + m) * Kdim + kk + ch * 8);
        }
#pragma unroll
        for (int i = 0; i < 2; ++i) {          // B tile 128x32 ([n][k] rows)
            int idx = tid + i * 256;
            int n = idx >> 2, ch = idx & 3;
            async_b128(Bs + n * LDB + ch * 8,
                       Bt + (size_t)(bn + n) * Kdim + kk + ch * 8);
        }
        WAIT_ASYNC();
        __syncthreads();

        v16bf afrag[4], bfrag[2];
#pragma unroll
        for (int ai = 0; ai < 4; ++ai) {
            const bf16_t* p = As + (wr * 64 + ai * 16 + r16) * LDA;
            afrag[ai] = cat16(*(const v8bf*)(p + g * 8), *(const v8bf*)(p + 16 + g * 8));
        }
#pragma unroll
        for (int bi = 0; bi < 2; ++bi) {
            const bf16_t* p = Bs + (wc * 32 + bi * 16 + r16) * LDB + g * 8;
            bfrag[bi] = cat16(*(const v8bf*)p, *(const v8bf*)(p + 16));
        }
#pragma unroll
        for (int ai = 0; ai < 4; ++ai)
#pragma unroll
            for (int bi = 0; bi < 2; ++bi)
                acc[ai][bi] = WMMA_BF16(afrag[ai], bfrag[bi], acc[ai][bi]);
    }

    __syncthreads();

    if (MODE == 0) {
#pragma unroll
        for (int ai = 0; ai < 4; ++ai)
#pragma unroll
            for (int bi = 0; bi < 2; ++bi) {
                int n = wc * 32 + bi * 16 + r16;
#pragma unroll
                for (int i = 0; i < 8; ++i)
                    Cs[(wr * 64 + ai * 16 + i + 8 * g) * LDC + n] = (bf16_t)acc[ai][bi][i];
            }
    } else {
#pragma unroll
        for (int ai = 0; ai < 4; ++ai)
#pragma unroll
            for (int bi = 0; bi < 2; ++bi) {
                int n  = wc * 32 + bi * 16 + r16;
                int m0 = wr * 64 + ai * 16 + 8 * g;
#pragma unroll
                for (int j = 0; j < 4; ++j)
                    *(unsigned int*)(Cs + n * LDC + m0 + 2 * j) =
                        pack_bf2(acc[ai][bi][2 * j], acc[ai][bi][2 * j + 1]);
            }
    }
    __syncthreads();

    if (MODE == 0) {
#pragma unroll
        for (int j = 0; j < 8; ++j) {
            int idx = tid + j * 256;
            int m = idx >> 4, ch = idx & 15;
            v8bf v = *(const v8bf*)(Cs + m * LDC + ch * 8);
            *(v8bf*)(out + (size_t)(bm + m) * N + bn + ch * 8) = v;
        }
    } else {
        const int bfix = bm >> 11, s0 = bm & 2047;
#pragma unroll
        for (int j = 0; j < 8; ++j) {
            int idx = tid + j * 256;
            int nl = idx >> 4, ch = idx & 15;
            int n = bn + nl, h = n >> 8, d = n & 255;
            v8bf v = *(const v8bf*)(Cs + nl * LDC + ch * 8);
            *(v8bf*)(out + (((size_t)(bfix * 4 + h) * 256 + d) << 11) + s0 + ch * 8) = v;
        }
    }
}

// ---------------------------------------------------------------------------
// Kernel 2: flash attention. grid = (S/128, H, B), block = 256 (8 waves).
// Wave 0 drives the TDM; wave 1 prefetches the next tile into L2.
// ---------------------------------------------------------------------------
__global__ __launch_bounds__(256) void attn_kernel(
    const bf16_t* __restrict__ Qb, const bf16_t* __restrict__ Kb,
    const bf16_t* __restrict__ VT, const int* __restrict__ sen_len,
    bf16_t* __restrict__ Zb)
{
    constexpr int LDK = 264, LDV = 40, LDP = 40, LDZ = 264;
    __shared__ __align__(16) bf16_t smem[23808];
    bf16_t* Ks = smem;                 // [key][d]   32 x 264
    bf16_t* Vs = smem + 8448;          // [d][key]  256 x 40
    bf16_t* Pw = smem + 18688;         // per-wave P scratch
    bf16_t* Zs = smem;                 // Z staging (after loop)

    const int tid  = threadIdx.x;
    const int lane = tid & 31, w = tid >> 5;
    const int g    = lane >> 4, r16 = lane & 15;
    const int qt = blockIdx.x, h = blockIdx.y, bb = blockIdx.z;
    const int q0 = qt * 128 + w * 16;
    const size_t rowbase = (size_t)bb * 2048;
    const int hcol = h * 256;

    v16bf qf[8];
    {
        const bf16_t* qp = Qb + (rowbase + q0 + r16) * 1024 + hcol;
#pragma unroll
        for (int dc = 0; dc < 8; ++dc) {
            const bf16_t* p = qp + dc * 32 + g * 8;
            qf[dc] = cat16(*(const v8bf*)p, *(const v8bf*)(p + 16));
        }
    }

    const v8f z8 = {0,0,0,0,0,0,0,0};
    v8f zacc[16];
#pragma unroll
    for (int t = 0; t < 16; ++t) zacc[t] = z8;
    float mrun[8], lrun[8];
#pragma unroll
    for (int i = 0; i < 8; ++i) { mrun[i] = -3.0e38f; lrun[i] = 0.0f; }

    int len = sen_len[bb];
    if (len < 1) len = 1;
    if (len > 2048) len = 2048;
    const int nkt = (len + 31) >> 5;

    const bf16_t* vt_base = VT + ((size_t)(bb * 4 + h) << 19);

    for (int kt = 0; kt < nkt; ++kt) {
        const int k0 = kt * 32;
        __syncthreads();                     // previous tiles fully consumed
#if HAVE_TDM
        if (w == 0) {
            tdm_load_2d_bf16((unsigned int)(size_t)Ks,
                             Kb + (rowbase + k0) * 1024 + hcol,
                             TDM_FLAGS_K, 256u, 32u, 256u, 32u, 1024ull);
            tdm_load_2d_bf16((unsigned int)(size_t)Vs,
                             vt_base + k0,
                             TDM_FLAGS_V, 32u, 256u, 32u, 256u, 2048ull);
            __builtin_amdgcn_s_wait_tensorcnt(0);
        }
        if (w == 1 && kt + 1 < nkt) {        // warm L2 for the next tile
            __builtin_prefetch(Kb + (rowbase + k0 + 32 + lane) * 1024 + hcol, 0, 1);
            __builtin_prefetch(vt_base + (((size_t)lane * 8) << 11) + k0 + 32, 0, 1);
        }
#else
#pragma unroll
        for (int i = 0; i < 4; ++i) {
            int idx = tid + i * 256;
            int kr = idx >> 5, ch = idx & 31;
            *(v8bf*)(Ks + kr * LDK + ch * 8) =
                *(const v8bf*)(Kb + (rowbase + k0 + kr) * 1024 + hcol + ch * 8);
        }
#pragma unroll
        for (int i = 0; i < 4; ++i) {
            int idx = tid + i * 256;
            int d = idx >> 2, ch = idx & 3;
            *(v8bf*)(Vs + d * LDV + ch * 8) =
                *(const v8bf*)(vt_base + ((size_t)d << 11) + k0 + ch * 8);
        }
#endif
        __syncthreads();                     // tiles visible to all waves

        v8f sc[2];
#pragma unroll
        for (int sub = 0; sub < 2; ++sub) {
            sc[sub] = z8;
#pragma unroll
            for (int dc = 0; dc < 8; ++dc) {
                const bf16_t* p = Ks + (sub * 16 + r16) * LDK + dc * 32 + g * 8;
                v16bf kb = cat16(*(const v8bf*)p, *(const v8bf*)(p + 16));
                sc[sub] = WMMA_BF16(qf[dc], kb, sc[sub]);
            }
        }

        float x0[8], x1[8];
        const bool ok0 = (k0 + r16) < len;
        const bool ok1 = (k0 + 16 + r16) < len;
#pragma unroll
        for (int i = 0; i < 8; ++i) {
            x0[i] = ok0 ? sc[0][i] * 0.03125f : -134217728.0f;   // PAD/32
            x1[i] = ok1 ? sc[1][i] * 0.03125f : -134217728.0f;
        }
        float corr[8];
#pragma unroll
        for (int i = 0; i < 8; ++i) {
            float t = fmaxf(x0[i], x1[i]);
            t = fmaxf(t, __shfl_xor(t, 1));
            t = fmaxf(t, __shfl_xor(t, 2));
            t = fmaxf(t, __shfl_xor(t, 4));
            t = fmaxf(t, __shfl_xor(t, 8));
            float mn = fmaxf(mrun[i], t);
            corr[i] = __expf(mrun[i] - mn);
            mrun[i] = mn;
            float p0 = __expf(x0[i] - mn);
            float p1 = __expf(x1[i] - mn);
            x0[i] = p0; x1[i] = p1;
            float rs = p0 + p1;
            rs += __shfl_xor(rs, 1);
            rs += __shfl_xor(rs, 2);
            rs += __shfl_xor(rs, 4);
            rs += __shfl_xor(rs, 8);
            lrun[i] = lrun[i] * corr[i] + rs;
        }
#pragma unroll
        for (int t = 0; t < 16; ++t)
#pragma unroll
            for (int i = 0; i < 8; ++i) zacc[t][i] *= corr[i];

        bf16_t* ps = Pw + w * 640;
#pragma unroll
        for (int i = 0; i < 8; ++i) {
            ps[(i + 8 * g) * LDP + r16]      = (bf16_t)x0[i];
            ps[(i + 8 * g) * LDP + 16 + r16] = (bf16_t)x1[i];
        }
        asm volatile("s_wait_dscnt 0" ::: "memory");
        v16bf pf;
        {
            const bf16_t* p = ps + r16 * LDP;
            pf = cat16(*(const v8bf*)(p + g * 8), *(const v8bf*)(p + 16 + g * 8));
        }
#pragma unroll
        for (int dsub = 0; dsub < 16; ++dsub) {
            const bf16_t* p = Vs + (dsub * 16 + r16) * LDV + g * 8;
            v16bf vb = cat16(*(const v8bf*)p, *(const v8bf*)(p + 16));
            zacc[dsub] = WMMA_BF16(pf, vb, zacc[dsub]);
        }
    }

    float inv[8];
#pragma unroll
    for (int i = 0; i < 8; ++i) inv[i] = 1.0f / lrun[i];

    __syncthreads();
    bf16_t* zs = Zs + w * (8 * LDZ);
#pragma unroll
    for (int p = 0; p < 2; ++p) {
        if (g == p) {
#pragma unroll
            for (int dsub = 0; dsub < 16; ++dsub)
#pragma unroll
                for (int i = 0; i < 8; ++i)
                    zs[i * LDZ + dsub * 16 + r16] = (bf16_t)(zacc[dsub][i] * inv[i]);
        }
        asm volatile("s_wait_dscnt 0" ::: "memory");
#pragma unroll
        for (int j = 0; j < 8; ++j) {
            int c = j * 32 + lane;
            int row = c >> 5, ch = c & 31;
            v8bf v = *(const v8bf*)(zs + row * LDZ + ch * 8);
            *(v8bf*)(Zb + (rowbase + q0 + p * 8 + row) * 1024 + hcol + ch * 8) = v;
        }
        asm volatile("s_wait_dscnt 0" ::: "memory");
    }
}

// ---------------------------------------------------------------------------
// Kernel 3: out = Z(bf16) @ WoT(bf16) + bo, fp32 out, async tiles.
// ---------------------------------------------------------------------------
__global__ __launch_bounds__(256) void out_gemm_kernel(
    const bf16_t* __restrict__ Zb, const bf16_t* __restrict__ WoT,
    const float* __restrict__ bo, float* __restrict__ out)
{
    constexpr int Kdim = 1024, N = 1024;
    constexpr int LDA = 40, LDB = 40;
    __shared__ __align__(16) bf16_t As[128 * LDA];
    __shared__ __align__(16) bf16_t Bs[128 * LDB];

    const int tid  = threadIdx.x;
    const int lane = tid & 31, w = tid >> 5;
    const int g    = lane >> 4, r16 = lane & 15;
    const int wr   = w >> 2,  wc  = w & 3;
    const int bm   = blockIdx.y * 128, bn = blockIdx.x * 128;

    const v8f z8 = {0,0,0,0,0,0,0,0};
    v8f acc[4][2];
#pragma unroll
    for (int a = 0; a < 4; ++a)
#pragma unroll
        for (int b = 0; b < 2; ++b) acc[a][b] = z8;

    for (int kk = 0; kk < Kdim; kk += 32) {
        __syncthreads();
#pragma unroll
        for (int i = 0; i < 2; ++i) {
            int idx = tid + i * 256;
            int m = idx >> 2, ch = idx & 3;
            async_b128(As + m * LDA + ch * 8,
                       Zb + (size_t)(bm + m) * Kdim + kk + ch * 8);
        }
#pragma unroll
        for (int i = 0; i < 2; ++i) {
            int idx = tid + i * 256;
            int n = idx >> 2, ch = idx & 3;
            async_b128(Bs + n * LDB + ch * 8,
                       WoT + (size_t)(bn + n) * Kdim + kk + ch * 8);
        }
        WAIT_ASYNC();
        __syncthreads();

        v16bf afrag[4], bfrag[2];
#pragma unroll
        for (int ai = 0; ai < 4; ++ai) {
            const bf16_t* p = As + (wr * 64 + ai * 16 + r16) * LDA;
            afrag[ai] = cat16(*(const v8bf*)(p + g * 8), *(const v8bf*)(p + 16 + g * 8));
        }
#pragma unroll
        for (int bi = 0; bi < 2; ++bi) {
            const bf16_t* p = Bs + (wc * 32 + bi * 16 + r16) * LDB + g * 8;
            bfrag[bi] = cat16(*(const v8bf*)p, *(const v8bf*)(p + 16));
        }
#pragma unroll
        for (int ai = 0; ai < 4; ++ai)
#pragma unroll
            for (int bi = 0; bi < 2; ++bi)
                acc[ai][bi] = WMMA_BF16(afrag[ai], bfrag[bi], acc[ai][bi]);
    }

#pragma unroll
    for (int ai = 0; ai < 4; ++ai)
#pragma unroll
        for (int bi = 0; bi < 2; ++bi) {
            int n = bn + wc * 32 + bi * 16 + r16;
            float bias = bo[n];
#pragma unroll
            for (int i = 0; i < 8; ++i) {
                int m = bm + wr * 64 + ai * 16 + i + 8 * g;
                out[(size_t)m * N + n] = acc[ai][bi][i] + bias;
            }
        }
}

// ---------------------------------------------------------------------------
// Launch
// ---------------------------------------------------------------------------
extern "C" void kernel_launch(void* const* d_in, const int* in_sizes, int n_in,
                              void* d_out, int out_size, void* d_ws, size_t ws_size,
                              hipStream_t stream) {
    (void)in_sizes; (void)n_in; (void)out_size; (void)ws_size;
    const float* X       = (const float*)d_in[0];
    const int*   sen_len = (const int*)d_in[1];
    const float* Wq      = (const float*)d_in[2];
    const float* Wk      = (const float*)d_in[3];
    const float* Wv      = (const float*)d_in[4];
    const float* Wo      = (const float*)d_in[5];
    const float* bo      = (const float*)d_in[6];
    float* out = (float*)d_out;

    const size_t elems = (size_t)16384 * 1024;   // 16M
    const size_t welem = (size_t)1024 * 1024;    // 1M
    bf16_t* Qb  = (bf16_t*)d_ws;
    bf16_t* Kb  = Qb + elems;
    bf16_t* VT  = Kb + elems;
    bf16_t* Zb  = VT + elems;
    bf16_t* Xb  = Zb + elems;
    bf16_t* WqT = Xb + elems;
    bf16_t* WkT = WqT + welem;
    bf16_t* WvT = WkT + welem;
    bf16_t* WoT = WvT + welem;

    dim3 blk(256);
    cvt_bf16_kernel<<<dim3(8192), blk, 0, stream>>>(X, Xb);
    wt_transpose_kernel<<<dim3(16, 16), blk, 0, stream>>>(Wq, WqT);
    wt_transpose_kernel<<<dim3(16, 16), blk, 0, stream>>>(Wk, WkT);
    wt_transpose_kernel<<<dim3(16, 16), blk, 0, stream>>>(Wv, WvT);
    wt_transpose_kernel<<<dim3(16, 16), blk, 0, stream>>>(Wo, WoT);
    proj_gemm_kernel<0><<<dim3(8, 128), blk, 0, stream>>>(Xb, WqT, Qb);
    proj_gemm_kernel<0><<<dim3(8, 128), blk, 0, stream>>>(Xb, WkT, Kb);
    proj_gemm_kernel<1><<<dim3(8, 128), blk, 0, stream>>>(Xb, WvT, VT);
    attn_kernel<<<dim3(16, 4, 8), blk, 0, stream>>>(Qb, Kb, VT, sen_len, Zb);
    out_gemm_kernel<<<dim3(8, 128), blk, 0, stream>>>(Zb, WoT, bo, out);
}
